// ParallelGatedConvBlock_86646670229723
// MI455X (gfx1250) — matrix operand order repaired and checked by reading, not compiled, not run on hardware
//
#include <hip/hip_runtime.h>
#include <hip/hip_bf16.h>

typedef unsigned short u16;
typedef __attribute__((ext_vector_type(16))) __bf16 v16bf;
typedef __attribute__((ext_vector_type(8)))  float  v8f;

// ---------------- constants ----------------
#define BB 2
#define LL 4096
#define DD 1024
#define SS 16
#define FFD 2688
#define D3 3072
#define MROWS (BB*LL)          // 8192
#define NCHUNK 64
#define CLEN 64                // NCHUNK*CLEN == LL

// ---------------- helpers ----------------
__device__ __forceinline__ u16 f2bf(float f) {
  union { float f; unsigned int u; } cv; cv.f = f;
  unsigned int u = cv.u;
  unsigned int r = u + 0x7FFFu + ((u >> 16) & 1u);
  return (u16)(r >> 16);
}

__device__ __forceinline__ float gelu_tanh(float x) {
  float x3 = x * x * x;
  return 0.5f * x * (1.0f + tanhf(0.7978845608028654f * (x + 0.044715f * x3)));
}

// ---------------- fp32 -> bf16 cast ----------------
__global__ __launch_bounds__(256) void cvt_f32_bf16_kernel(
    const float* __restrict__ in, u16* __restrict__ out, int n) {
  int i = blockIdx.x * 256 + threadIdx.x;
  if (i < n) out[i] = f2bf(in[i]);
}

// ---------------- RMSNorm (row of 1024) -> bf16 ----------------
__global__ __launch_bounds__(256) void rmsnorm_bf16_kernel(
    const float* __restrict__ x, const float* __restrict__ w,
    u16* __restrict__ out) {
  const int row = blockIdx.x;
  const int tid = threadIdx.x;
  const float* px = x + (size_t)row * DD;
  float v[4]; float s = 0.f;
#pragma unroll
  for (int j = 0; j < 4; ++j) { v[j] = px[tid + j * 256]; s = fmaf(v[j], v[j], s); }
#pragma unroll
  for (int m = 16; m > 0; m >>= 1) s += __shfl_xor(s, m, 32);
  __shared__ float red[8];
  if ((tid & 31) == 0) red[tid >> 5] = s;
  __syncthreads();
  float tot = 0.f;
#pragma unroll
  for (int i = 0; i < 8; ++i) tot += red[i];
  const float scale = rsqrtf(tot * (1.0f / (float)DD) + 1e-6f);
  u16* po = out + (size_t)row * DD;
#pragma unroll
  for (int j = 0; j < 4; ++j) po[tid + j * 256] = f2bf(v[j] * scale * w[tid + j * 256]);
}

// ---------------- WMMA bf16 GEMM: C[M,N] = A[M,K] @ W[N,K]^T ----------------
// Triple-buffered LDS fed by async global->LDS DMA (ASYNCcnt), 2-stage-deep
// prefetch: s_wait_asynccnt 6 retires only the oldest in-flight stage.
// MODE 0: C fp32 = acc
// MODE 1: C fp32 = acc + R
// MODE 2: CB bf16 = gelu(R) * acc
union Frag32 { uint4 q[2]; v16bf v; };

template<int MODE>
__global__ __launch_bounds__(256) void gemm_bf16_wmma(
    const u16* __restrict__ A, const u16* __restrict__ W,
    float* __restrict__ C, u16* __restrict__ CB,
    const float* __restrict__ R, int Ndim, int Kdim) {
  __shared__ alignas(16) u16 lA[3][64 * 72];
  __shared__ alignas(16) u16 lB[3][128 * 72];

  const int tid   = threadIdx.x;
  const int wid   = tid >> 5;
  const int lane  = tid & 31;
  const int lhalf = lane >> 4;
  const int lrow  = lane & 15;
  const int wm    = wid & 1;     // 2 wave rows (32 M each)
  const int wn    = wid >> 1;    // 4 wave cols (32 N each)
  const int mBase = blockIdx.y * 64;
  const int nBase = blockIdx.x * 128;

  v8f acc[2][2] = {};

  const int ar = tid >> 2, ac = (tid & 3) << 4;   // A fill: 64 rows x 64 halves, 32B/thread
  const int br = tid >> 1, bc = (tid & 1) << 5;   // B fill: 128 rows x 64 halves, 64B/thread
  const u16* gA = A + (size_t)(mBase + ar) * Kdim + ac;
  const u16* gB = W + (size_t)(nBase + br) * Kdim + bc;
  const unsigned laOff0 = (unsigned)(size_t)&lA[0][ar * 72 + ac];
  const unsigned laOff1 = (unsigned)(size_t)&lA[1][ar * 72 + ac];
  const unsigned laOff2 = (unsigned)(size_t)&lA[2][ar * 72 + ac];
  const unsigned lbOff0 = (unsigned)(size_t)&lB[0][br * 72 + bc];
  const unsigned lbOff1 = (unsigned)(size_t)&lB[1][br * 72 + bc];
  const unsigned lbOff2 = (unsigned)(size_t)&lB[2][br * 72 + bc];

  const int nIter = Kdim >> 6;

  // async DMA fill of one LDS stage: 6 async instructions per thread.
  // (instruction offset applies to BOTH the global and the LDS address,
  //  and the 16B sub-chunks are contiguous in both address spaces)
  auto fillAsync = [&](int buf, int k0) {
    const unsigned la = (buf == 0) ? laOff0 : ((buf == 1) ? laOff1 : laOff2);
    const unsigned lb = (buf == 0) ? lbOff0 : ((buf == 1) ? lbOff1 : lbOff2);
    unsigned long long ga = (unsigned long long)(gA + k0);
    asm volatile(
        "global_load_async_to_lds_b128 %0, %1, off\n\t"
        "global_load_async_to_lds_b128 %0, %1, off offset:16"
        :: "v"(la), "v"(ga) : "memory");
    unsigned long long gb = (unsigned long long)(gB + k0);
    asm volatile(
        "global_load_async_to_lds_b128 %0, %1, off\n\t"
        "global_load_async_to_lds_b128 %0, %1, off offset:16\n\t"
        "global_load_async_to_lds_b128 %0, %1, off offset:32\n\t"
        "global_load_async_to_lds_b128 %0, %1, off offset:48"
        :: "v"(lb), "v"(gb) : "memory");
  };

  fillAsync(0, 0);
  if (nIter > 1) fillAsync(1, 64);

  int buf = 0;
  for (int it = 0; it < nIter; ++it) {
    // retire the oldest in-flight stage; newest (it+1) may stay outstanding
    if (it + 1 < nIter) asm volatile("s_wait_asynccnt 0x6" ::: "memory");
    else                asm volatile("s_wait_asynccnt 0x0" ::: "memory");
    __syncthreads();

    // prefetch stage it+2 into the buffer whose reads finished at iteration it-1
    if (it + 2 < nIter) {
      int nb = buf + 2; if (nb >= 3) nb -= 3;
      fillAsync(nb, (it + 2) << 6);
    }

    const u16* sA = lA[buf];
    const u16* sB = lB[buf];
#pragma unroll
    for (int kk = 0; kk < 64; kk += 32) {
      Frag32 a[2], b[2];
#pragma unroll
      for (int f = 0; f < 2; ++f) {
        // A 16x32 layout: lanes 0-15 K in {0..7,16..23}, lanes 16-31 K in {8..15,24..31}
        const u16* base = &sA[(wm * 32 + f * 16 + lrow) * 72 + kk + lhalf * 8];
        a[f].q[0] = *(const uint4*)(base);
        a[f].q[1] = *(const uint4*)(base + 16);
      }
#pragma unroll
      for (int g = 0; g < 2; ++g) {
        // B 32x16: column = lane&15, lanes 0-15 K in {0..7,16..23}, lanes 16-31 K in {8..15,24..31}
        const u16* base = &sB[(wn * 32 + g * 16 + lrow) * 72 + kk + lhalf * 8];
        b[g].q[0] = *(const uint4*)(base);
        b[g].q[1] = *(const uint4*)(base + 16);
      }
#pragma unroll
      for (int f = 0; f < 2; ++f)
#pragma unroll
        for (int g = 0; g < 2; ++g)
          acc[f][g] = __builtin_amdgcn_wmma_f32_16x16x32_bf16(
              false, a[f].v, false, b[g].v, (short)0, acc[f][g], false, false);
    }

    ++buf; if (buf >= 3) buf = 0;
  }

#pragma unroll
  for (int f = 0; f < 2; ++f) {
    const int row0 = mBase + wm * 32 + f * 16 + lhalf * 8;
#pragma unroll
    for (int g = 0; g < 2; ++g) {
      const int col = nBase + wn * 32 + g * 16 + lrow;
#pragma unroll
      for (int i = 0; i < 8; ++i) {
        const size_t idx = (size_t)(row0 + i) * Ndim + col;
        float v = acc[f][g][i];
        if (MODE == 1) v += R[idx];
        if (MODE == 2) {
          CB[idx] = f2bf(gelu_tanh(R[idx]) * v);
        } else {
          C[idx] = v;
        }
      }
    }
  }
}

// ---------------- causal depthwise FIR (K=3) + split + gate prep ----------------
__global__ __launch_bounds__(256) void fir_gate_kernel(
    const float* __restrict__ z, const float* __restrict__ sw,
    const float* __restrict__ sb, float* __restrict__ x2,
    float* __restrict__ xv) {
  const long long idx = (long long)blockIdx.x * 256 + threadIdx.x;  // < B*L*D
  const int d = (int)(idx & (DD - 1));
  const int t = (int)((idx >> 10) & (LL - 1));
  const int b = (int)(idx >> 22);
  float r[3];
#pragma unroll
  for (int j = 0; j < 3; ++j) {
    const int c = j * DD + d;
    float acc = sb[c];
#pragma unroll
    for (int k = 0; k < 3; ++k) {
      int tt = t + k - 2;
      if (tt >= 0) acc = fmaf(sw[c * 3 + k], z[((size_t)(b * LL + tt)) * D3 + c], acc);
    }
    r[j] = acc;
  }
  const size_t o = (size_t)(b * LL + t) * DD + d;
  x2[o] = r[0];
  xv[o] = r[1] * r[2];
}

// ---------------- chunked SSM scan (order-16 diagonal recurrence) ----------------
__global__ __launch_bounds__(256) void ssm_passA(
    const float* __restrict__ xv, const float* __restrict__ lp,
    float* __restrict__ cend) {
  const int tid = blockIdx.x * 256 + threadIdx.x;   // < B*NCHUNK*D
  const int c = tid & (DD - 1);
  const int chunk = (tid >> 10) & (NCHUNK - 1);
  const int b = tid >> 16;
  float decay[SS], st[SS];
#pragma unroll
  for (int s = 0; s < SS; ++s) { decay[s] = expf(lp[c * SS + s]); st[s] = 0.f; }
  const float* px = xv + ((size_t)(b * LL + chunk * CLEN)) * DD + c;
  for (int t = 0; t < CLEN; ++t) {
    float x = px[(size_t)t * DD];
#pragma unroll
    for (int s = 0; s < SS; ++s) st[s] = fmaf(decay[s], st[s], x);
  }
  float* pe = cend + ((size_t)((b * DD + c) * NCHUNK + chunk)) * SS;
#pragma unroll
  for (int s = 0; s < SS; ++s) pe[s] = st[s];
}

__global__ __launch_bounds__(256) void ssm_passB(
    const float* __restrict__ lp, const float* __restrict__ cend,
    float* __restrict__ cinit) {
  const int tid = blockIdx.x * 256 + threadIdx.x;   // < B*D
  const int c = tid & (DD - 1);
  const int b = tid >> 10;
  float dpow[SS], st[SS];
#pragma unroll
  for (int s = 0; s < SS; ++s) { dpow[s] = expf(lp[c * SS + s] * (float)CLEN); st[s] = 0.f; }
  const size_t base = ((size_t)(b * DD + c)) * NCHUNK * SS;
  for (int ch = 0; ch < NCHUNK; ++ch) {
#pragma unroll
    for (int s = 0; s < SS; ++s) cinit[base + ch * SS + s] = st[s];
#pragma unroll
    for (int s = 0; s < SS; ++s) st[s] = fmaf(dpow[s], st[s], cend[base + ch * SS + s]);
  }
}

__global__ __launch_bounds__(256) void ssm_passC(
    const float* __restrict__ xv, const float* __restrict__ x2,
    const float* __restrict__ lp, const float* __restrict__ res,
    const float* __restrict__ dskip, const float* __restrict__ cinit,
    u16* __restrict__ yact) {
  const int tid = blockIdx.x * 256 + threadIdx.x;
  const int c = tid & (DD - 1);
  const int chunk = (tid >> 10) & (NCHUNK - 1);
  const int b = tid >> 16;
  float decay[SS], st[SS], rr[SS];
#pragma unroll
  for (int s = 0; s < SS; ++s) {
    decay[s] = expf(lp[c * SS + s]);
    rr[s] = res[c * SS + s];
  }
  const float* pi = cinit + ((size_t)((b * DD + c) * NCHUNK + chunk)) * SS;
#pragma unroll
  for (int s = 0; s < SS; ++s) st[s] = pi[s];
  const float dsk = dskip[c];
  const size_t rb = ((size_t)(b * LL + chunk * CLEN)) * DD + c;
  for (int t = 0; t < CLEN; ++t) {
    const size_t o = rb + (size_t)t * DD;
    float x = xv[o];
    float acc = 0.f;
#pragma unroll
    for (int s = 0; s < SS; ++s) { st[s] = fmaf(decay[s], st[s], x); acc = fmaf(rr[s], st[s], acc); }
    float y = (acc + x * dsk) * x2[o];
    yact[o] = f2bf(y);
  }
}

// ---------------- workspace layout ----------------
static constexpr size_t ALN = 256;
static constexpr size_t alup(size_t x) { return (x + ALN - 1) & ~(ALN - 1); }
static constexpr size_t SZ_WIN   = alup((size_t)D3 * DD * 2);
static constexpr size_t SZ_WOUT  = alup((size_t)DD * DD * 2);
static constexpr size_t SZ_WL1   = alup((size_t)FFD * DD * 2);
static constexpr size_t SZ_WL2   = SZ_WL1;
static constexpr size_t SZ_WL3   = alup((size_t)DD * FFD * 2);
static constexpr size_t SZ_XN    = alup((size_t)MROWS * DD * 2);
static constexpr size_t SZ_Z     = alup((size_t)MROWS * D3 * 4);   // also hosts g1 fp32
static constexpr size_t SZ_X2    = alup((size_t)MROWS * DD * 4);
static constexpr size_t SZ_XV    = SZ_X2;                          // x2+xv also host hB bf16
static constexpr size_t SZ_CST   = alup((size_t)BB * DD * NCHUNK * SS * 4);

static constexpr size_t O_WIN  = 0;
static constexpr size_t O_WOUT = O_WIN + SZ_WIN;
static constexpr size_t O_WL1  = O_WOUT + SZ_WOUT;
static constexpr size_t O_WL2  = O_WL1 + SZ_WL1;
static constexpr size_t O_WL3  = O_WL2 + SZ_WL2;
static constexpr size_t O_XN   = O_WL3 + SZ_WL3;
static constexpr size_t O_Z    = O_XN + SZ_XN;
static constexpr size_t O_X2   = O_Z + SZ_Z;
static constexpr size_t O_XV   = O_X2 + SZ_X2;
static constexpr size_t O_CEND = O_XV + SZ_XV;
static constexpr size_t O_CINI = O_CEND + SZ_CST;
static constexpr size_t O_YACT = O_CINI + SZ_CST;
static constexpr size_t O_Y2   = O_YACT + SZ_XN;
static constexpr size_t O_ZM   = O_Y2 + SZ_X2;
static constexpr size_t O_G1   = O_Z;    // alias: z dead after FIR
static constexpr size_t O_HB   = O_X2;   // alias: x2/xv dead after pass C

extern "C" void kernel_launch(void* const* d_in, const int* in_sizes, int n_in,
                              void* d_out, int out_size, void* d_ws, size_t ws_size,
                              hipStream_t stream) {
  const float* u    = (const float*)d_in[0];
  const float* prw  = (const float*)d_in[1];
  const float* win  = (const float*)d_in[2];
  const float* sw   = (const float*)d_in[3];
  const float* sb   = (const float*)d_in[4];
  const float* lp   = (const float*)d_in[5];
  const float* res  = (const float*)d_in[6];
  const float* dsk  = (const float*)d_in[7];
  const float* wout = (const float*)d_in[8];
  const float* pnw  = (const float*)d_in[9];
  const float* wl1  = (const float*)d_in[10];
  const float* wl2  = (const float*)d_in[11];
  const float* wl3  = (const float*)d_in[12];
  float* out = (float*)d_out;

  char* ws = (char*)d_ws;
  u16*   wInB  = (u16*)(ws + O_WIN);
  u16*   wOutB = (u16*)(ws + O_WOUT);
  u16*   wL1B  = (u16*)(ws + O_WL1);
  u16*   wL2B  = (u16*)(ws + O_WL2);
  u16*   wL3B  = (u16*)(ws + O_WL3);
  u16*   xnB   = (u16*)(ws + O_XN);
  float* zF    = (float*)(ws + O_Z);
  float* x2F   = (float*)(ws + O_X2);
  float* xvF   = (float*)(ws + O_XV);
  float* cendF = (float*)(ws + O_CEND);
  float* ciniF = (float*)(ws + O_CINI);
  u16*   yactB = (u16*)(ws + O_YACT);
  float* y2F   = (float*)(ws + O_Y2);
  u16*   zmB   = (u16*)(ws + O_ZM);
  float* g1F   = (float*)(ws + O_G1);
  u16*   hB    = (u16*)(ws + O_HB);

  // weights -> bf16
  cvt_f32_bf16_kernel<<<(D3 * DD + 255) / 256, 256, 0, stream>>>(win, wInB, D3 * DD);
  cvt_f32_bf16_kernel<<<(DD * DD + 255) / 256, 256, 0, stream>>>(wout, wOutB, DD * DD);
  cvt_f32_bf16_kernel<<<(FFD * DD + 255) / 256, 256, 0, stream>>>(wl1, wL1B, FFD * DD);
  cvt_f32_bf16_kernel<<<(FFD * DD + 255) / 256, 256, 0, stream>>>(wl2, wL2B, FFD * DD);
  cvt_f32_bf16_kernel<<<(DD * FFD + 255) / 256, 256, 0, stream>>>(wl3, wL3B, DD * FFD);

  // pre-norm -> bf16
  rmsnorm_bf16_kernel<<<MROWS, 256, 0, stream>>>(u, prw, xnB);

  // z = xnorm @ in_proj^T      [8192 x 3072]
  gemm_bf16_wmma<0><<<dim3(D3 / 128, MROWS / 64), 256, 0, stream>>>(
      xnB, wInB, zF, (u16*)nullptr, (const float*)nullptr, D3, DD);

  // FIR + split + xv = x1*v
  fir_gate_kernel<<<(BB * LL * DD) / 256, 256, 0, stream>>>(zF, sw, sb, x2F, xvF);

  // chunked SSM scan
  ssm_passA<<<(BB * NCHUNK * DD) / 256, 256, 0, stream>>>(xvF, lp, cendF);
  ssm_passB<<<(BB * DD) / 256, 256, 0, stream>>>(lp, cendF, ciniF);
  ssm_passC<<<(BB * NCHUNK * DD) / 256, 256, 0, stream>>>(xvF, x2F, lp, res, dsk, ciniF, yactB);

  // y2 = yact @ out_proj^T + u   [8192 x 1024] fp32
  gemm_bf16_wmma<1><<<dim3(DD / 128, MROWS / 64), 256, 0, stream>>>(
      yactB, wOutB, y2F, (u16*)nullptr, u, DD, DD);

  // post-norm -> bf16
  rmsnorm_bf16_kernel<<<MROWS, 256, 0, stream>>>(y2F, pnw, zmB);

  // g1 = zm @ l1^T               [8192 x 2688] fp32
  gemm_bf16_wmma<0><<<dim3(FFD / 128, MROWS / 64), 256, 0, stream>>>(
      zmB, wL1B, g1F, (u16*)nullptr, (const float*)nullptr, FFD, DD);

  // h = gelu(g1) * (zm @ l2^T)   [8192 x 2688] bf16
  gemm_bf16_wmma<2><<<dim3(FFD / 128, MROWS / 64), 256, 0, stream>>>(
      zmB, wL2B, (float*)nullptr, hB, g1F, FFD, DD);

  // out = h @ l3^T + y2          [8192 x 1024] fp32
  gemm_bf16_wmma<1><<<dim3(DD / 128, MROWS / 64), 256, 0, stream>>>(
      hB, wL3B, out, (u16*)nullptr, y2F, DD, FFD);
}
